// EchoChamberV2_75892072121009
// MI455X (gfx1250) — compile-verified, matrix operand order checked by reading, not run on hardware
//
#include <hip/hip_runtime.h>
#include <math.h>

#define AS1 __attribute__((address_space(1)))
#define AS3 __attribute__((address_space(3)))

typedef __attribute__((ext_vector_type(2)))  float    v2f;
typedef __attribute__((ext_vector_type(8)))  float    v8f;
typedef __attribute__((ext_vector_type(16))) _Float16 v16h;
typedef int v4i_gcc __attribute__((vector_size(16)));  // matches builtin param type

#define LUT_N    4096
#define LUT_MASK 4095
#define PHI_F    1.61803398874989484820f
#define TWO_PI_F 6.28318530717958647692f
#define STEP_F   (TWO_PI_F / (float)LUT_N)
#define SCALE_F  ((float)LUT_N / TWO_PI_F)
#define NTHREADS 512

#if defined(__has_builtin)
#  if __has_builtin(__builtin_amdgcn_global_load_async_to_lds_b128)
#    define ECHO_ASYNC 1
#  endif
#  if __has_builtin(__builtin_amdgcn_s_wait_asynccnt)
#    define ECHO_WAIT_ASYNC_BUILTIN 1
#  endif
#  if __has_builtin(__builtin_amdgcn_wmma_f32_16x16x4_f32)
#    define ECHO_WMMA_F32 1
#  elif __has_builtin(__builtin_amdgcn_wmma_f32_16x16x32_f16)
#    define ECHO_WMMA_F16 1
#  endif
#endif

__device__ __forceinline__ int lut_index(float theta) {
  // jnp.round == round-half-even == rintf (RNE); jnp.mod(n, 4096) == n & 4095
  return ((int)rintf(theta * SCALE_F)) & LUT_MASK;
}

// Wave32 cross-lane sum via WMMA. Input: per-lane value v. Output: per-lane
// partials that are zero except on lanes with (lane&15)==0; the sum of the
// returned values over the wave equals sum(v) over all 32 lanes.
// A = {v, 0}: row-sum over K collapses to v[m] + v[m+16] no matter which K
// slot each VGPR holds. B = column-0 mask (rows of B stripe N across lanes,
// so lane&15==0 covers column 0 of every K row) -> D[m,0] = v[m]+v[m+16],
// D[m,n!=0] = 0. Summing the 8 D VGPRs per lane leaves partials on lanes 0/16.
__device__ __forceinline__ float wave_reduce_partial(float v, float colmask) {
#if defined(ECHO_WMMA_F32)
  v2f a;  a[0] = v;        a[1] = 0.0f;
  v2f bm; bm[0] = colmask; bm[1] = colmask;
  v8f c = {};
  v8f d = __builtin_amdgcn_wmma_f32_16x16x4_f32(
      /*neg_a=*/false, a, /*neg_b=*/false, bm,
      /*c_mod=*/(short)0, c, /*reuse_a=*/false, /*reuse_b=*/false);
  return d[0] + d[1] + d[2] + d[3] + d[4] + d[5] + d[6] + d[7];
#elif defined(ECHO_WMMA_F16)
  v16h a = {};
  a[0] = (_Float16)v;
  _Float16 cm = (_Float16)colmask;
  v16h bm;
#pragma unroll
  for (int i = 0; i < 16; ++i) bm[i] = cm;
  v8f c = {};
  v8f d = __builtin_amdgcn_wmma_f32_16x16x32_f16(
      false, a, false, bm, (short)0, c, false, false);
  return d[0] + d[1] + d[2] + d[3] + d[4] + d[5] + d[6] + d[7];
#else
  // Plain fallback: half-wave shuffle reduce; lanes 0 and 16 hold the sums.
  for (int off = 1; off < 16; off <<= 1) v += __shfl_down(v, off, 32);
  return v;
#endif
}

// One workgroup per row b (B = 4096 rows, D = 4096 cols). x row staged in LDS
// once via async global->LDS copies, reused by both phases.
__global__ __launch_bounds__(NTHREADS) void echo_chamber_v2_kernel(
    const float* __restrict__ x_real, const float* __restrict__ x_imag,
    const float* __restrict__ t,
    const float* __restrict__ trig_r, const float* __restrict__ trig_i,
    const float* __restrict__ w_q,    const float* __restrict__ b_q,
    const float* __restrict__ w_out,  const float* __restrict__ b_out,
    const float* __restrict__ beta,
    float* __restrict__ out_real, float* __restrict__ out_imag,
    int D, float invH) {
  __shared__ __align__(16) float s_xr[4096];
  __shared__ __align__(16) float s_xi[4096];
  __shared__ float s_acc[2];
  __shared__ float s_mag;

  const int b   = blockIdx.x;
  const int tid = threadIdx.x;

  if (tid == 0) { s_acc[0] = 0.0f; s_acc[1] = 0.0f; }

  // ---- stage row of x_real / x_imag into LDS (async DMA path) ----
  {
    const float4* gxr4 = reinterpret_cast<const float4*>(x_real + (size_t)b * (size_t)D);
    const float4* gxi4 = reinterpret_cast<const float4*>(x_imag + (size_t)b * (size_t)D);
    float4* lxr4 = reinterpret_cast<float4*>(s_xr);
    float4* lxi4 = reinterpret_cast<float4*>(s_xi);
    const int nvec = D >> 2;  // 1024 float4 per array
#if defined(ECHO_ASYNC)
    for (int f = tid; f < nvec; f += NTHREADS) {
      __builtin_amdgcn_global_load_async_to_lds_b128(
          (AS1 v4i_gcc*)(gxr4 + f), (AS3 v4i_gcc*)(lxr4 + f), 0, 0);
      __builtin_amdgcn_global_load_async_to_lds_b128(
          (AS1 v4i_gcc*)(gxi4 + f), (AS3 v4i_gcc*)(lxi4 + f), 0, 0);
    }
#  if defined(ECHO_WAIT_ASYNC_BUILTIN)
    __builtin_amdgcn_s_wait_asynccnt(0);
#  else
    asm volatile("s_wait_asynccnt 0" ::: "memory");
#  endif
#else
    for (int f = tid; f < nvec; f += NTHREADS) {
      lxr4[f] = gxr4[f];
      lxi4[f] = gxi4[f];
    }
#endif
  }
  __syncthreads();

  const float tphi = t[b] * PHI_F;

  // ---- phase 1: per-row interference reduction ----
  float accr = 0.0f, acci = 0.0f;
  for (int d = tid; d < D; d += NTHREADS) {
    float xr = s_xr[d], xi = s_xi[d];
    float wq = w_q[d],  bq = b_q[d];
    float inv_wl = 1.0f / (1.0f + fabsf(wq));
    float thr = xr * inv_wl + bq + tphi;
    float thi = xi * inv_wl + bq + tphi;
    int ir = lut_index(thr);
    int ii = lut_index(thi);
    // cosr*cosi - sinr*sini = cos(a+b); cosr*sini + sinr*cosi = sin(a+b)
    float ang = (float)((ir + ii) & LUT_MASK) * STEP_F;
    float qr = __cosf(ang);
    float qi = __sinf(ang);
    float tr = trig_r[d], ti = trig_i[d];
    accr = fmaf(qr, tr, fmaf(qi, ti, accr));
    acci = fmaf(qr, ti, fmaf(-qi, tr, acci));
  }

  // wave-level reduce via WMMA, then block combine via LDS float atomics
  const float colmask = ((tid & 15) == 0) ? 1.0f : 0.0f;
  float pr = wave_reduce_partial(accr, colmask);
  float pi = wave_reduce_partial(acci, colmask);
  if ((tid & 15) == 0) {
    atomicAdd(&s_acc[0], pr);
    atomicAdd(&s_acc[1], pi);
  }
  __syncthreads();
  if (tid == 0) {
    float trt = s_acc[0] * invH;  // mean over heads == (1/H) * sum over all d
    float tit = s_acc[1] * invH;
    s_mag = sqrtf(trt * trt + tit * tit + 1e-8f);
  }
  __syncthreads();
  const float mag = s_mag;

  // ---- phase 2: memory write-scale + Euler projection ----
  for (int d = tid; d < D; d += NTHREADS) {
    float xr = s_xr[d], xi = s_xi[d];
    float wo = w_out[d], bo = b_out[d], be = beta[d];
    float w_eff    = 1.0f / (1.0f + fabsf(wo));   // also 1/wavelength
    float beta_eff = 1.0f / (1.0f + fabsf(be));
    float decay = fminf(__expf(-beta_eff * w_eff), 0.9999f);
    float sc = mag * (w_eff * (1.0f - decay));    // int_mag * write_scale
    float thr = (xr * sc) * w_eff + bo + tphi;    // mem/wavelength + b + tφ
    float thi = (xi * sc) * w_eff + bo + tphi;
    int ir = lut_index(thr);
    int ii = lut_index(thi);
    float ang = (float)((ir + ii) & LUT_MASK) * STEP_F;
    size_t o = (size_t)b * (size_t)D + (size_t)d;
    out_real[o] = __cosf(ang);
    out_imag[o] = __sinf(ang);
  }
}

extern "C" void kernel_launch(void* const* d_in, const int* in_sizes, int n_in,
                              void* d_out, int out_size, void* d_ws, size_t ws_size,
                              hipStream_t stream) {
  (void)n_in; (void)out_size; (void)d_ws; (void)ws_size;
  const float* x_real = (const float*)d_in[0];
  const float* x_imag = (const float*)d_in[1];
  const float* t      = (const float*)d_in[2];
  const float* trig_r = (const float*)d_in[3];
  const float* trig_i = (const float*)d_in[4];
  const float* w_q    = (const float*)d_in[5];
  const float* b_q    = (const float*)d_in[6];
  const float* w_out  = (const float*)d_in[7];
  const float* b_out  = (const float*)d_in[8];
  const float* beta   = (const float*)d_in[9];

  const int B = in_sizes[2];  // t has B elements
  const int D = in_sizes[7];  // w_out has D elements
  const float invH = 1.0f / 16.0f;  // H = 16 per reference setup

  float* out_real = (float*)d_out;
  float* out_imag = out_real + (size_t)B * (size_t)D;

  echo_chamber_v2_kernel<<<dim3(B), dim3(NTHREADS), 0, stream>>>(
      x_real, x_imag, t, trig_r, trig_i, w_q, b_q, w_out, b_out, beta,
      out_real, out_imag, D, invH);
}
